// VQLayer_30442728194287
// MI455X (gfx1250) — matrix operand order repaired and checked by reading, not compiled, not run on hardware
//
#include <hip/hip_runtime.h>

typedef __attribute__((ext_vector_type(16))) _Float16 v16h;
typedef __attribute__((ext_vector_type(8)))  _Float16 v8h;
typedef __attribute__((ext_vector_type(8)))  float    v8f;
typedef __attribute__((ext_vector_type(4)))  unsigned int v4u;
typedef __attribute__((ext_vector_type(8)))  int      v8i;
typedef __attribute__((ext_vector_type(4)))  int      v4i;

#define N_LAT   131072
#define NPROTO  1024
#define DIM     64
#define PROW    72          // padded proto row stride in halfs (conflict-free LDS)
#define NBLOCKS 512
#define WAVES   8
#define ITERS   2           // NBLOCKS*WAVES*ITERS*16 == N_LAT

// d_ws layout (floats unless noted):
//   [0..1023]    softmax bins accumulator
//   [1024]       SSE accumulator
//   [1056..2079] ||p||^2 per prototype
//   byte 8448    f16 prototypes, row-major [1024][64] (131072 bytes)
#define WS_SSE        1024
#define WS_Q          1056
#define WS_PRO16_BYTE 8448

// ---- WMMA operand builders ------------------------------------------------
// A (16x32 f16) per ISA 7.12.2: lane row = lane%16; lane<16 holds K {0-7,16-23},
// lane>=16 holds K {8-15,24-31}. Source is f32 row-major in global.
__device__ __forceinline__ v16h load_a_half(const float* __restrict__ xrow, int k0, int kbase) {
  const float4 f0 = *(const float4*)(xrow + kbase + k0);
  const float4 f1 = *(const float4*)(xrow + kbase + k0 + 4);
  const float4 f2 = *(const float4*)(xrow + kbase + k0 + 16);
  const float4 f3 = *(const float4*)(xrow + kbase + k0 + 20);
  v16h a;
  a[0]=(_Float16)f0.x; a[1]=(_Float16)f0.y; a[2]=(_Float16)f0.z; a[3]=(_Float16)f0.w;
  a[4]=(_Float16)f1.x; a[5]=(_Float16)f1.y; a[6]=(_Float16)f1.z; a[7]=(_Float16)f1.w;
  a[8]=(_Float16)f2.x; a[9]=(_Float16)f2.y; a[10]=(_Float16)f2.z; a[11]=(_Float16)f2.w;
  a[12]=(_Float16)f3.x; a[13]=(_Float16)f3.y; a[14]=(_Float16)f3.z; a[15]=(_Float16)f3.w;
  return a;
}

// B (32x16 f16): lane column = lane%16 (proto row), lanes 0-15 hold K {0-7,16-23},
// lanes 16-31 hold K {8-15,24-31} (matches A layout).
__device__ __forceinline__ v16h load_b(const _Float16* prow, int kofs) {
  const v8h lo = *(const v8h*)(prow + kofs);
  const v8h hi = *(const v8h*)(prow + kofs + 16);
  v16h b;
#pragma unroll
  for (int i = 0; i < 8; ++i) { b[i] = lo[i]; b[8 + i] = hi[i]; }
  return b;
}

__device__ __forceinline__ v8f wmma_f16(v16h a, v16h b, v8f c) {
  return __builtin_amdgcn_wmma_f32_16x16x32_f16(false, a, false, b, (short)0, c, false, false);
}

// ---- kernels --------------------------------------------------------------
// One thread per prototype: zero accumulators, f32->f16 convert, ||p||^2.
__global__ __launch_bounds__(256) void vq_prep(const float* __restrict__ pro,
                                               float* __restrict__ ws) {
  const int p = blockIdx.x * 256 + threadIdx.x;   // 0..1023
  ws[p] = 0.0f;                                   // bins
  if (p == 0) ws[WS_SSE] = 0.0f;

  const float* pr = pro + p * DIM;
  _Float16* dst = (_Float16*)((char*)ws + WS_PRO16_BYTE) + p * DIM;
  float q = 0.0f;
#pragma unroll
  for (int c = 0; c < 8; ++c) {
    const float4 u = *(const float4*)(pr + c * 8);
    const float4 v = *(const float4*)(pr + c * 8 + 4);
    q += u.x*u.x + u.y*u.y + u.z*u.z + u.w*u.w
       + v.x*v.x + v.y*v.y + v.z*v.z + v.w*v.w;
    v8h h;
    h[0]=(_Float16)u.x; h[1]=(_Float16)u.y; h[2]=(_Float16)u.z; h[3]=(_Float16)u.w;
    h[4]=(_Float16)v.x; h[5]=(_Float16)v.y; h[6]=(_Float16)v.z; h[7]=(_Float16)v.w;
    *(v8h*)(dst + c * 8) = h;
  }
  ws[WS_Q + p] = q;
}

__global__ __launch_bounds__(256) void vq_main(const float* __restrict__ lat,
                                               const float* __restrict__ pro,
                                               float* __restrict__ outq,
                                               float* __restrict__ ws) {
  extern __shared__ char smem[];
  _Float16* shp = (_Float16*)smem;                         // [NPROTO][PROW] f16 (TDM dest)
  float*    shq = (float*)(smem + NPROTO * PROW * 2);      // ||p||^2
  float*    shb = shq + NPROTO;                            // softmax bins

  const int tid  = threadIdx.x;
  const int lane = tid & 31;
  const int wave = tid >> 5;

  // Wave 0: DMA the f16 prototype array global->LDS via the Tensor Data Mover.
  // 2D tensor of DWORDs: dim0=32 (128B row), dim1=1024 rows; hardware pads
  // +4 DWORDs after every 32 DWORDs -> LDS row stride 144B == PROW halfs.
  if (wave == 0) {
    const unsigned long long ga =
        (unsigned long long)(uintptr_t)((const char*)ws + WS_PRO16_BYTE);
    const unsigned lds_base = (unsigned)(uintptr_t)(void*)smem;  // LDS byte offset
    v4u g0;
    g0.x = 1u;                                   // count=1 valid descriptor
    g0.y = lds_base;                             // lds_addr
    g0.z = (unsigned)ga;                         // global_addr[31:0]
    g0.w = (unsigned)((ga >> 32) & 0x1FFFFFFu)   // global_addr[56:32]
         | (2u << 30);                           // type = 2 ("image")
    v8i g1;
    g1[0] = (int)((2u << 16)                     // data_size = 4 bytes
                | (1u << 20)                     // pad_enable
                | (4u << 22)                     // pad_interval: 32 DWORDs
                | (3u << 25));                   // pad_amount: 4 DWORDs
    g1[1] = (int)(32u << 16);                    // tensor_dim0 = 32 (low 16 bits)
    g1[2] = (int)(1024u << 16);                  // tensor_dim1 = 1024 (low 16 bits)
    g1[3] = (int)(32u << 16);                    // tile_dim0 = 32
    g1[4] = 1024;                                // tile_dim1 = 1024 (tile_dim2 = 0)
    g1[5] = 32;                                  // tensor_dim0_stride = 32
    g1[6] = 0;
    g1[7] = 0;
    const v4i gz4 = {0, 0, 0, 0};                // groups 2/3 unused (2D tensor)
    const v8i gz8 = {0, 0, 0, 0, 0, 0, 0, 0};
    __builtin_amdgcn_tensor_load_to_lds(g0, g1, gz4, gz4, gz8, 0);
  }

  // Overlap with the DMA: stage ||p||^2 and zero the bins.
  for (int p = tid; p < NPROTO; p += 256) {
    shq[p] = ws[WS_Q + p];
    shb[p] = 0.0f;
  }

  if (wave == 0) __builtin_amdgcn_s_wait_tensorcnt(0);
  __syncthreads();

  const int col = lane & 15;             // A row / B proto column within tile
  const int ka  = (lane >> 4) * 8;       // A K sub-offset
  const int kb  = (lane >> 4) * 8;       // B K sub-offset
  float sse_local = 0.0f;

  for (int it = 0; it < ITERS; ++it) {
    const int strip = it * (NBLOCKS * WAVES) + blockIdx.x * WAVES + wave;
    const int row0  = strip * 16;
    const float* xrow = lat + (size_t)(row0 + col) * DIM;
    const v16h a0 = load_a_half(xrow, ka, 0);
    const v16h a1 = load_a_half(xrow, ka, 32);

    float m[8], ssum[8];
    int   idx[8];
#pragma unroll
    for (int i = 0; i < 8; ++i) { m[i] = -1e30f; ssum[i] = 0.0f; idx[i] = 0; }

    // Pass 1: argmax(a) == argmin(dist), and per-row sum of exp(a).
    for (int t = 0; t < 64; ++t) {
      const int p = t * 16 + col;
      const _Float16* prow = shp + p * PROW;
      v8f acc = {};
      acc = wmma_f16(a0, load_b(prow, kb), acc);
      acc = wmma_f16(a1, load_b(prow, kb + 32), acc);
      const float qv = shq[p];
#pragma unroll
      for (int i = 0; i < 8; ++i) {
        const float av = 2.0f * acc[i] - qv;       // -dist + ||x||^2
        ssum[i] += __expf(av);                     // |av| ~ 0.1: no shift needed
        if (av > m[i]) { m[i] = av; idx[i] = p; }
      }
    }

    // Reduce (max,idx,sum) across the 16 lanes holding each row.
#pragma unroll
    for (int mask = 1; mask <= 8; mask <<= 1) {
#pragma unroll
      for (int i = 0; i < 8; ++i) {
        const float om = __shfl_xor(m[i], mask, 32);
        const int   oi = __shfl_xor(idx[i], mask, 32);
        const float os = __shfl_xor(ssum[i], mask, 32);
        ssum[i] += os;
        if (om > m[i] || (om == m[i] && oi < idx[i])) { m[i] = om; idx[i] = oi; }
      }
    }
    float inv[8];
#pragma unroll
    for (int i = 0; i < 8; ++i) inv[i] = 1.0f / ssum[i];

    // Pass 2: accumulate normalized soft assignment into per-proto bins.
    for (int t = 0; t < 64; ++t) {
      const int p = t * 16 + col;
      const _Float16* prow = shp + p * PROW;
      v8f acc = {};
      acc = wmma_f16(a0, load_b(prow, kb), acc);
      acc = wmma_f16(a1, load_b(prow, kb + 32), acc);
      const float qv = shq[p];
      float tsum = 0.0f;
#pragma unroll
      for (int i = 0; i < 8; ++i) tsum += __expf(2.0f * acc[i] - qv) * inv[i];
      tsum += __shfl_xor(tsum, 16, 32);            // rows 0-7 + rows 8-15
      if (lane < 16) atomicAdd(&shb[p], tsum);
    }

    // Gather quantized rows from exact f32 prototypes; accumulate SSE.
#pragma unroll
    for (int r = 0; r < 16; ++r) {
      const int src  = (r >> 3) << 4;              // lane 0 or 16 holds row r
      const int pidx = __shfl(idx[r & 7], src, 32);
      const float2 pv = *(const float2*)(pro + (size_t)pidx * DIM + 2 * lane);
      const float2 xv = *(const float2*)(lat + (size_t)(row0 + r) * DIM + 2 * lane);
      *(float2*)(outq + (size_t)(row0 + r) * DIM + 2 * lane) = pv;
      const float dx = pv.x - xv.x, dy = pv.y - xv.y;
      sse_local += dx * dx + dy * dy;
    }
  }

  __syncthreads();
  for (int p = tid; p < NPROTO; p += 256) atomicAdd(&ws[p], shb[p]);
  atomicAdd(&ws[WS_SSE], sse_local);
}

__global__ __launch_bounds__(256) void vq_fin(const float* __restrict__ ws,
                                              float* __restrict__ out) {
  __shared__ float rA[256], rB[256];
  const int tid = threadIdx.x;
  float A = 0.0f, B = 0.0f;
  for (int p = tid; p < NPROTO; p += 256) {
    const float sp = ws[p] * (1.0f / (float)N_LAT) + 1e-8f;
    A += sp;
    B += sp * logf(sp);
  }
  rA[tid] = A; rB[tid] = B;
  __syncthreads();
  for (int s = 128; s > 0; s >>= 1) {
    if (tid < s) { rA[tid] += rA[tid + s]; rB[tid] += rB[tid + s]; }
    __syncthreads();
  }
  if (tid == 0) {
    const float At = rA[0], Bt = rB[0];
    const size_t base = (size_t)N_LAT * DIM;
    out[base]     = ws[WS_SSE] * (1.25f / ((float)N_LAT * (float)DIM)); // vq_loss
    out[base + 1] = logf(At) - Bt / At;                                 // entropy
  }
}

extern "C" void kernel_launch(void* const* d_in, const int* in_sizes, int n_in,
                              void* d_out, int out_size, void* d_ws, size_t ws_size,
                              hipStream_t stream) {
  const float* lat = (const float*)d_in[0];
  const float* pro = (const float*)d_in[1];
  float* out = (float*)d_out;
  float* ws  = (float*)d_ws;

  vq_prep<<<4, 256, 0, stream>>>(pro, ws);
  const size_t smem = (size_t)NPROTO * PROW * 2 + (size_t)NPROTO * 4 * 2; // 155,648 B
  vq_main<<<NBLOCKS, 256, smem, stream>>>(lat, pro, out, ws);
  vq_fin<<<1, 256, 0, stream>>>(ws, out);
}
